// MemoryModule_17154099380435
// MI455X (gfx1250) — compile-verified, matrix operand order checked by reading, not run on hardware
//
#include <hip/hip_runtime.h>
#include <hip/hip_bf16.h>

// ---------------------------------------------------------------------------
// Channel-attention "MemoryModule" for MI455X (gfx1250), wave32 + WMMA bf16.
//   B=8, C=512, N=W*H=4096
//   stage 1: k = key_w*x + key_b ; v = value_w*x + value_b   (bf16 out)
//   stage 2: energy[b] = k[b] (CxN) * q^T (NxC), split-K=4   (f32 partials)
//   stage 3: att_t[b][c][d] = softmax_d(sum partials)[d][c]  (bf16, transposed)
//   stage 4: y = gamma * (att_t[b] (CxC) * v[b] (CxN)) + x   (f32 out)
// GEMMs: 128x128 macro-tile, BK=64 (2 WMMA K-steps), double-buffered LDS,
// one barrier per K-iteration, v_wmma_f32_16x16x32_bf16.
// ---------------------------------------------------------------------------

typedef __attribute__((ext_vector_type(16))) __bf16 v16bf;
typedef __attribute__((ext_vector_type(8)))  float  v8f;

#define C_DIM 512
#define NPIX  4096
#define BK    64
#define LDT   72          // BK + 8 ushort pad: 144B row stride, conflict-free
#define TSZ   (128 * LDT) // one LDS tile (ushorts)

union FragBF { v16bf v; unsigned int u[8]; };

static __device__ __forceinline__ unsigned short f2bf(float f) {
    unsigned int u = __float_as_uint(f);
    unsigned int r = u + 0x7FFFu + ((u >> 16) & 1u);   // round-to-nearest-even
    return (unsigned short)(r >> 16);
}

// One loader serves A (LDS [m][k]) and B (LDS stored transposed as [n][k]):
// lane 0-15 -> row = lane, kh=0 ; lane 16-31 -> same row, kh=8.
// VGPR r: K pair base = 2r (r<4) else 2r+8, + kh. Matches ISA 16-bit A layout.
static __device__ __forceinline__ v16bf load_frag(const unsigned short* S,
                                                  int rbase, int ko, int lane) {
    FragBF f;
    const unsigned short* p = S + (size_t)(rbase + (lane & 15)) * LDT + ko;
    const int kh = (lane >> 4) << 3;
#pragma unroll
    for (int r = 0; r < 8; ++r) {
        int kb = ((r < 4) ? (2 * r) : (2 * r + 8)) + kh;
        f.u[r] = *(const unsigned int*)(p + kb);       // two packed bf16
    }
    return f.v;
}

static __device__ __forceinline__ v8f wmma_bf16(v16bf a, v16bf b, v8f c) {
    return __builtin_amdgcn_wmma_f32_16x16x32_bf16(
        false, a, false, b, (short)0, c, false, false);
}

// 2 K-steps x (2x4) fragment tiles = 16 WMMA per call
static __device__ __forceinline__ void compute_tile(
    const unsigned short* As, const unsigned short* Bs,
    int m0, int n0, int lane, v8f (&acc)[2][4])
{
#pragma unroll
    for (int ks = 0; ks < BK; ks += 32) {
        v16bf af[2], bfr[4];
#pragma unroll
        for (int mt = 0; mt < 2; ++mt) af[mt]  = load_frag(As, m0 + mt * 16, ks, lane);
#pragma unroll
        for (int nt = 0; nt < 4; ++nt) bfr[nt] = load_frag(Bs, n0 + nt * 16, ks, lane);
#pragma unroll
        for (int mt = 0; mt < 2; ++mt)
#pragma unroll
            for (int nt = 0; nt < 4; ++nt)
                acc[mt][nt] = wmma_bf16(af[mt], bfr[nt], acc[mt][nt]);
    }
}

// ---- LDS staging: 256 threads, 128x64 = 8192 elems (32 contiguous/thread) --
// A tile: src row-major [m][k] -> As[m][k]
static __device__ __forceinline__ void stage_A_f32(unsigned short* dst,
        const float* __restrict__ src, int ld, int tid) {
    const int r = tid >> 1, cb = (tid & 1) << 5;
#pragma unroll
    for (int i = 0; i < 32; ++i)
        dst[r * LDT + cb + i] = f2bf(src[(size_t)r * ld + cb + i]);
}
static __device__ __forceinline__ void stage_A_bf16(unsigned short* dst,
        const unsigned short* __restrict__ src, int ld, int tid) {
    const int r = tid >> 1, cb = (tid & 1) << 5;
#pragma unroll
    for (int i = 0; i < 32; ++i)
        dst[r * LDT + cb + i] = src[(size_t)r * ld + cb + i];
}
// B tile: src row-major [k][n] (64x128) -> Bs[n][k] (transpose on LDS write)
static __device__ __forceinline__ void stage_B_f32(unsigned short* dst,
        const float* __restrict__ src, int ld, int tid) {
    const int k = tid >> 2, nb = (tid & 3) << 5;
#pragma unroll
    for (int i = 0; i < 32; ++i)
        dst[(nb + i) * LDT + k] = f2bf(src[(size_t)k * ld + nb + i]);
}
static __device__ __forceinline__ void stage_B_bf16(unsigned short* dst,
        const unsigned short* __restrict__ src, int ld, int tid) {
    const int k = tid >> 2, nb = (tid & 3) << 5;
#pragma unroll
    for (int i = 0; i < 32; ++i)
        dst[(nb + i) * LDT + k] = src[(size_t)k * ld + nb + i];
}
// B tile from transposed source: src row-major [n][k] (q) -> Bs[n][k]
static __device__ __forceinline__ void stage_Bt_f32(unsigned short* dst,
        const float* __restrict__ src, int ld, int tid) {
    const int n = tid >> 1, kb2 = (tid & 1) << 5;
#pragma unroll
    for (int i = 0; i < 32; ++i)
        dst[n * LDT + kb2 + i] = f2bf(src[(size_t)n * ld + kb2 + i]);
}

// ---------------------------------------------------------------------------
// Stage 1: 1x1 conv (key & value fused via blockIdx.z parity)
// grid (NPIX/128=32, C/128=4, 2*B=16)
// ---------------------------------------------------------------------------
__global__ __launch_bounds__(256) void conv1x1_kernel(
    const float* __restrict__ x,
    const float* __restrict__ key_w, const float* __restrict__ key_b,
    const float* __restrict__ value_w, const float* __restrict__ value_b,
    unsigned short* __restrict__ k_out, unsigned short* __restrict__ v_out)
{
    __shared__ unsigned short As[2 * TSZ];
    __shared__ unsigned short Bs[2 * TSZ];
    const int tid = threadIdx.x, lane = tid & 31, wave = tid >> 5;
    const int m0 = (wave >> 1) * 32, n0 = (wave & 1) * 64;
    const int bn = blockIdx.x * 128, bm = blockIdx.y * 128;
    const int batch = blockIdx.z >> 1, which = blockIdx.z & 1;
    const float* W    = which ? value_w : key_w;
    const float* bias = which ? value_b : key_b;
    unsigned short* out = (which ? v_out : k_out) + (size_t)batch * C_DIM * NPIX;
    const float* xb = x + (size_t)batch * C_DIM * NPIX;

    v8f acc[2][4];
#pragma unroll
    for (int mt = 0; mt < 2; ++mt)
#pragma unroll
        for (int nt = 0; nt < 4; ++nt) acc[mt][nt] = (v8f)0.0f;

    const int NIT = C_DIM / BK;   // 8
    stage_A_f32(As, W + (size_t)bm * C_DIM, C_DIM, tid);
    stage_B_f32(Bs, xb + bn, NPIX, tid);
    __syncthreads();
    int buf = 0;
    for (int it = 0; it < NIT; ++it) {
        if (it + 1 < NIT) {
            const int kn = (it + 1) * BK;
            stage_A_f32(As + (buf ^ 1) * TSZ, W + (size_t)bm * C_DIM + kn, C_DIM, tid);
            stage_B_f32(Bs + (buf ^ 1) * TSZ, xb + (size_t)kn * NPIX + bn, NPIX, tid);
            __builtin_prefetch(xb + (size_t)(kn + BK) * NPIX + bn, 0, 1);
        }
        compute_tile(As + buf * TSZ, Bs + buf * TSZ, m0, n0, lane, acc);
        __syncthreads();
        buf ^= 1;
    }
#pragma unroll
    for (int mt = 0; mt < 2; ++mt)
#pragma unroll
        for (int nt = 0; nt < 4; ++nt)
#pragma unroll
            for (int e = 0; e < 8; ++e) {
                int Mg = bm + m0 + mt * 16 + e + ((lane >> 4) << 3);
                int Ng = bn + n0 + nt * 16 + (lane & 15);
                out[(size_t)Mg * NPIX + Ng] = f2bf(acc[mt][nt][e] + bias[Mg]);
            }
}

// ---------------------------------------------------------------------------
// Stage 2: energy partials, split-K=4:
//   epart[s][b][c][d] = sum_{n in chunk s} k[b,c,n] * q[d,n]
// grid (C/128=4, C/128=4, 4*B=32)
// ---------------------------------------------------------------------------
__global__ __launch_bounds__(256) void energy_kernel(
    const unsigned short* __restrict__ kmat,
    const float* __restrict__ q,
    float* __restrict__ epart)
{
    __shared__ unsigned short As[2 * TSZ];
    __shared__ unsigned short Bs[2 * TSZ];
    const int tid = threadIdx.x, lane = tid & 31, wave = tid >> 5;
    const int m0 = (wave >> 1) * 32, n0 = (wave & 1) * 64;
    const int bn = blockIdx.x * 128, bm = blockIdx.y * 128;
    const int batch = blockIdx.z >> 2, ksl = blockIdx.z & 3;
    const int kbase = ksl * (NPIX / 4);           // 1024-wide K chunk
    const unsigned short* ka = kmat + (size_t)batch * C_DIM * NPIX;
    float* eb = epart + ((size_t)ksl * 8 + batch) * C_DIM * C_DIM;

    v8f acc[2][4];
#pragma unroll
    for (int mt = 0; mt < 2; ++mt)
#pragma unroll
        for (int nt = 0; nt < 4; ++nt) acc[mt][nt] = (v8f)0.0f;

    const int NIT = (NPIX / 4) / BK;   // 16
    stage_A_bf16(As, ka + (size_t)bm * NPIX + kbase, NPIX, tid);
    stage_Bt_f32(Bs, q + (size_t)bn * NPIX + kbase, NPIX, tid);
    __syncthreads();
    int buf = 0;
    for (int it = 0; it < NIT; ++it) {
        if (it + 1 < NIT) {
            const int kn = kbase + (it + 1) * BK;
            stage_A_bf16(As + (buf ^ 1) * TSZ, ka + (size_t)bm * NPIX + kn, NPIX, tid);
            stage_Bt_f32(Bs + (buf ^ 1) * TSZ, q + (size_t)bn * NPIX + kn, NPIX, tid);
            __builtin_prefetch(ka + (size_t)bm * NPIX + kn + BK, 0, 1);
        }
        compute_tile(As + buf * TSZ, Bs + buf * TSZ, m0, n0, lane, acc);
        __syncthreads();
        buf ^= 1;
    }
#pragma unroll
    for (int mt = 0; mt < 2; ++mt)
#pragma unroll
        for (int nt = 0; nt < 4; ++nt)
#pragma unroll
            for (int e = 0; e < 8; ++e) {
                int Mg = bm + m0 + mt * 16 + e + ((lane >> 4) << 3);
                int Ng = bn + n0 + nt * 16 + (lane & 15);
                eb[(size_t)Mg * C_DIM + Ng] = acc[mt][nt][e];
            }
}

// ---------------------------------------------------------------------------
// Stage 3: sum 4 partials + row softmax over d, written transposed:
//   att_t[b][d][r] = softmax(row r)[d]
// grid (B*C = 4096), 256 threads, 2 elems/thread
// ---------------------------------------------------------------------------
__global__ __launch_bounds__(256) void softmax_t_kernel(
    const float* __restrict__ epart, unsigned short* __restrict__ att_t)
{
    __shared__ float red[256];
    const int tid = threadIdx.x;
    const int b = blockIdx.x >> 9, r = blockIdx.x & 511;
    const size_t PART = (size_t)8 * C_DIM * C_DIM;
    const size_t rowoff = ((size_t)b * C_DIM + r) * C_DIM;

    float e0 = 0.0f, e1 = 0.0f;
#pragma unroll
    for (int s = 0; s < 4; ++s) {
        e0 += epart[s * PART + rowoff + tid];
        e1 += epart[s * PART + rowoff + tid + 256];
    }

    red[tid] = fmaxf(e0, e1);
    __syncthreads();
    for (int s = 128; s > 0; s >>= 1) {
        if (tid < s) red[tid] = fmaxf(red[tid], red[tid + s]);
        __syncthreads();
    }
    const float mx = red[0];
    __syncthreads();

    float x0 = __expf(e0 - mx), x1 = __expf(e1 - mx);
    red[tid] = x0 + x1;
    __syncthreads();
    for (int s = 128; s > 0; s >>= 1) {
        if (tid < s) red[tid] += red[tid + s];
        __syncthreads();
    }
    const float inv = 1.0f / red[0];

    unsigned short* at = att_t + (size_t)b * C_DIM * C_DIM;
    at[(size_t)tid * C_DIM + r]         = f2bf(x0 * inv);
    at[(size_t)(tid + 256) * C_DIM + r] = f2bf(x1 * inv);
}

// ---------------------------------------------------------------------------
// Stage 4: y[b,c,n] = gamma * sum_d att_t[b,c,d] * v[b,d,n] + x[b,c,n]
// grid (NPIX/128=32, C/128=4, B=8)
// ---------------------------------------------------------------------------
__global__ __launch_bounds__(256) void out_kernel(
    const unsigned short* __restrict__ att_t,
    const unsigned short* __restrict__ vmat,
    const float* __restrict__ x,
    const float* __restrict__ gamma,
    float* __restrict__ y)
{
    __shared__ unsigned short As[2 * TSZ];
    __shared__ unsigned short Bs[2 * TSZ];
    const int tid = threadIdx.x, lane = tid & 31, wave = tid >> 5;
    const int m0 = (wave >> 1) * 32, n0 = (wave & 1) * 64;
    const int bn = blockIdx.x * 128, bm = blockIdx.y * 128;
    const int batch = blockIdx.z;
    const unsigned short* ab = att_t + (size_t)batch * C_DIM * C_DIM;
    const unsigned short* vb = vmat + (size_t)batch * C_DIM * NPIX;
    const float* xb = x + (size_t)batch * C_DIM * NPIX;
    float* yb = y + (size_t)batch * C_DIM * NPIX;
    const float g = gamma[0];

    v8f acc[2][4];
#pragma unroll
    for (int mt = 0; mt < 2; ++mt)
#pragma unroll
        for (int nt = 0; nt < 4; ++nt) acc[mt][nt] = (v8f)0.0f;

    const int NIT = C_DIM / BK;   // 8
    stage_A_bf16(As, ab + (size_t)bm * C_DIM, C_DIM, tid);
    stage_B_bf16(Bs, vb + bn, NPIX, tid);
    __syncthreads();
    int buf = 0;
    for (int it = 0; it < NIT; ++it) {
        if (it + 1 < NIT) {
            const int kn = (it + 1) * BK;
            stage_A_bf16(As + (buf ^ 1) * TSZ, ab + (size_t)bm * C_DIM + kn, C_DIM, tid);
            stage_B_bf16(Bs + (buf ^ 1) * TSZ, vb + (size_t)kn * NPIX + bn, NPIX, tid);
            __builtin_prefetch(vb + (size_t)(kn + BK) * NPIX + bn, 0, 1);
        }
        compute_tile(As + buf * TSZ, Bs + buf * TSZ, m0, n0, lane, acc);
        __syncthreads();
        buf ^= 1;
    }
#pragma unroll
    for (int mt = 0; mt < 2; ++mt)
#pragma unroll
        for (int nt = 0; nt < 4; ++nt)
#pragma unroll
            for (int e = 0; e < 8; ++e) {
                int Mg = bm + m0 + mt * 16 + e + ((lane >> 4) << 3);
                int Ng = bn + n0 + nt * 16 + (lane & 15);
                size_t idx = (size_t)Mg * NPIX + Ng;
                yb[idx] = g * acc[mt][nt][e] + xb[idx];
            }
}

// ---------------------------------------------------------------------------
extern "C" void kernel_launch(void* const* d_in, const int* in_sizes, int n_in,
                              void* d_out, int out_size, void* d_ws, size_t ws_size,
                              hipStream_t stream) {
    const float* x       = (const float*)d_in[0];   // 8*512*64*64
    const float* q       = (const float*)d_in[1];   // 512*64*64
    const float* key_w   = (const float*)d_in[2];   // 512*512
    const float* key_b   = (const float*)d_in[3];   // 512
    const float* value_w = (const float*)d_in[4];   // 512*512
    const float* value_b = (const float*)d_in[5];   // 512
    const float* gamma   = (const float*)d_in[6];   // 1
    float* y = (float*)d_out;

    const size_t BCN = (size_t)8 * C_DIM * NPIX;    // 16M elems
    const size_t BCC = (size_t)8 * C_DIM * C_DIM;   // 2M elems
    // ws: kb(32MB bf16) | vb(32MB bf16) | epart(4x8MB f32) | att_t(4MB bf16)
    unsigned short* kb    = (unsigned short*)d_ws;
    unsigned short* vb    = kb + BCN;
    float*          ep    = (float*)(vb + BCN);
    unsigned short* att_t = (unsigned short*)(ep + 4 * BCC);

    conv1x1_kernel<<<dim3(32, 4, 16), 256, 0, stream>>>(
        x, key_w, key_b, value_w, value_b, kb, vb);
    energy_kernel<<<dim3(4, 4, 32), 256, 0, stream>>>(kb, q, ep);
    softmax_t_kernel<<<dim3(4096), 256, 0, stream>>>(ep, att_t);
    out_kernel<<<dim3(32, 4, 8), 256, 0, stream>>>(att_t, vb, x, gamma, y);
}
